// LSTM_70300024700981
// MI455X (gfx1250) — matrix operand order repaired and checked by reading, not compiled, not run on hardware
//
#include <hip/hip_runtime.h>
#include <hip/hip_bf16.h>

// ---------------------------------------------------------------------------
// LSTM (B=512, T=1024, F=128, H=512) + MLP head, for MI455X (gfx1250).
//
// Strategy:
//  * prep_kernel  : pack [W_ih | W_hh] into bf16 WMMA B-fragment tiles
//                   (K=640 = 128(x) + 512(h)), combine biases. ~2.5MB -> L2.
//  * lstm_kernel  : persistent, 32 workgroups (one per 16 batch rows),
//                   8 waves each. Per step: stage x_t (f32->bf16) into an LDS
//                   A-panel, run v_wmma_f32_16x16x32_bf16 over K=640 for the
//                   i/f/g/o gate columns, fuse sigmoid/tanh/cell update in
//                   registers, write new h (bf16) into the double-buffered
//                   LDS A-panel for the next step. c state lives in VGPRs.
//  * head_kernel  : relu(hT @ W1^T + b1) @ W2^T + b2 -> [512] floats.
// ---------------------------------------------------------------------------

typedef __attribute__((ext_vector_type(16))) __bf16 v16bf;
typedef __attribute__((ext_vector_type(8)))  float  v8f;

#define BB   512
#define TT   1024
#define FF   128
#define HH   512
#define G4   2048               // 4*H
#define KK   640                // F + H
#define KT   20                 // K tiles of 32
#define NTG  128                // 2048/16 gate column tiles
#define BTILE 16                // batch rows per workgroup

// workspace layout (bytes)
#define HT_OFF   0                                   // [512*512] f32 hT
#define WP_OFF   (512*512*4)                         // packed weights bf16
#define WP_BYTES ((size_t)NTG*KT*512*2)              // 128*20 tiles * 1KB
#define BIAS_OFF (WP_OFF + (int)WP_BYTES)            // [2048] f32 combined bias

// LDS A-panel layout (per workgroup):
//   x region : tiles kt=0..3   at  kt*1024
//   h ping   : tiles kt=4..19  at  4096 + ph*16384 + (kt-4)*1024
// A-fragment tile (16M x 32K bf16, 1KB): element (M, kl) at
//   lane = (M&15) + 16*(kl>>4), slot j = kl&15  ->  lane*32 + j*2
#define LDS_BYTES (4096 + 2*16384)

__device__ __forceinline__ float sigf(float x) {
    return 1.0f / (1.0f + __expf(-x));
}
__device__ __forceinline__ float tanh_fast(float x) {
    // tanh(x) = 2*sigmoid(2x) - 1
    return 2.0f / (1.0f + __expf(-2.0f * x)) - 1.0f;
}

// ---------------------------------------------------------------------------
// Pack weights into WMMA B-fragment tiles + combine biases.
// B-fragment tile (32K x 16N bf16): lane l holds N col (g = nt*16 + (l&15)),
// slot j holds K row (k = kt*32 + 16*(l>>4) + j). Tile order: (nt*KT + kt), 512 elems.
// ---------------------------------------------------------------------------
__global__ __launch_bounds__(256) void prep_kernel(
    const float* __restrict__ Wih, const float* __restrict__ Whh,
    const float* __restrict__ bih, const float* __restrict__ bhh,
    char* __restrict__ ws)
{
    __bf16* wp   = (__bf16*)(ws + WP_OFF);
    float*  bias = (float*)(ws + BIAS_OFF);

    size_t idx    = (size_t)blockIdx.x * blockDim.x + threadIdx.x;
    size_t nthr   = (size_t)gridDim.x * blockDim.x;

    if (idx < G4) bias[idx] = bih[idx] + bhh[idx];

    const size_t total = (size_t)NTG * KT * 512;
    for (size_t p = idx; p < total; p += nthr) {
        int tile   = (int)(p >> 9);
        int within = (int)(p & 511);
        int lane   = within >> 4;          // N col = lane&15, K half = lane>>4
        int j      = within & 15;          // K within half
        int nt     = tile / KT;
        int kt     = tile - nt * KT;
        int g      = nt * 16 + (lane & 15); // gate row 0..2047
        int k      = kt * 32 + ((lane >> 4) << 4) + j; // K 0..639
        float v = (k < FF) ? Wih[(size_t)g * FF + k]
                           : Whh[(size_t)g * HH + (k - FF)];
        wp[p] = (__bf16)v;
    }
}

// ---------------------------------------------------------------------------
// Persistent LSTM recurrence.
// grid = 32 (B/16), block = 256 (8 waves). Wave w owns hidden n-tiles
// nt_h = w*4+q (q=0..3) -> gate tiles nt_h, nt_h+32, nt_h+64, nt_h+96.
// C/D tile layout: lane l -> N = l&15, VGPR r -> M = r + 8*(l>>4).
// ---------------------------------------------------------------------------
__global__ __launch_bounds__(256) void lstm_kernel(
    const float*  __restrict__ x,       // [B, T, F]
    const __bf16* __restrict__ wp,      // packed weights
    const float*  __restrict__ bias,    // [2048] combined
    float*        __restrict__ hT)      // [B, H] out
{
    extern __shared__ char smem[];

    const int tid  = threadIdx.x;
    const int wave = tid >> 5;
    const int lane = tid & 31;
    const int b0   = blockIdx.x * BTILE;

    // zero LDS (h ping/pong start at 0 == h_0 = 0)
    for (int i = tid; i < LDS_BYTES / 4; i += 256)
        ((unsigned*)smem)[i] = 0u;

    // cell state: 4 n-tiles x 8 rows per lane
    float c_reg[4][8];
#pragma unroll
    for (int q = 0; q < 4; ++q)
#pragma unroll
        for (int r = 0; r < 8; ++r) c_reg[q][r] = 0.0f;

    __syncthreads();

    const int ncol = lane & 15;     // N within tile
    const int mhi  = (lane >> 4);   // 0 or 1 -> M = r or r+8

    for (int t = 0; t < TT; ++t) {
        const int ph = t & 1;

        // ---- stage x_t: 16 rows x 128 K -> bf16 A-panel tiles kt=0..3 ----
#pragma unroll
        for (int i = 0; i < 8; ++i) {
            int e      = tid + i * 256;        // 0..2047
            int kt     = e >> 9;
            int within = e & 511;
            int la     = within >> 4;
            int j      = within & 15;
            int M      = la & 15;
            int kl     = ((la >> 4) << 4) + j; // 0..31
            float v = x[((size_t)(b0 + M) * TT + t) * FF + (kt * 32 + kl)];
            *(__bf16*)(smem + kt * 1024 + la * 32 + j * 2) = (__bf16)v;
        }
        __syncthreads();

        const char* hsrc = smem + 4096 + ph * 16384;
        char*       hdst = smem + 4096 + (1 - ph) * 16384;

#pragma unroll 1
        for (int q = 0; q < 4; ++q) {
            const int nt_h = wave * 4 + q;          // 0..31
            const int col  = nt_h * 16 + ncol;      // hidden column 0..511

            v8f acc_i, acc_f, acc_g, acc_o;
            {
                float bi = bias[col];
                float bf = bias[HH + col];
                float bg = bias[2 * HH + col];
                float bo = bias[3 * HH + col];
#pragma unroll
                for (int r = 0; r < 8; ++r) {
                    acc_i[r] = bi; acc_f[r] = bf;
                    acc_g[r] = bg; acc_o[r] = bo;
                }
            }

            // K loop over 640 in steps of 32; A-fragment reused by 4 gates
#pragma unroll 4
            for (int kt = 0; kt < KT; ++kt) {
                const char* ap = (kt < 4)
                    ? (smem + kt * 1024 + lane * 32)
                    : (hsrc + (kt - 4) * 1024 + lane * 32);
                v16bf a = *(const v16bf*)ap;

                const __bf16* bpi = wp + ((size_t)(nt_h       ) * KT + kt) * 512 + lane * 16;
                const __bf16* bpf = wp + ((size_t)(nt_h +  32 ) * KT + kt) * 512 + lane * 16;
                const __bf16* bpg = wp + ((size_t)(nt_h +  64 ) * KT + kt) * 512 + lane * 16;
                const __bf16* bpo = wp + ((size_t)(nt_h +  96 ) * KT + kt) * 512 + lane * 16;

                v16bf bi = *(const v16bf*)bpi;
                v16bf bf = *(const v16bf*)bpf;
                v16bf bg = *(const v16bf*)bpg;
                v16bf bo = *(const v16bf*)bpo;

                acc_i = __builtin_amdgcn_wmma_f32_16x16x32_bf16(false, a, false, bi, (short)0, acc_i, false, false);
                acc_f = __builtin_amdgcn_wmma_f32_16x16x32_bf16(false, a, false, bf, (short)0, acc_f, false, false);
                acc_g = __builtin_amdgcn_wmma_f32_16x16x32_bf16(false, a, false, bg, (short)0, acc_g, false, false);
                acc_o = __builtin_amdgcn_wmma_f32_16x16x32_bf16(false, a, false, bo, (short)0, acc_o, false, false);
            }

            // ---- fused gate math + cell update, write h into pong panel ----
            const int k   = FF + col;          // A-panel K index of this column
            const int kt2 = k >> 5;
            const int kl  = k & 31;
            const int j   = kl & 15;
            const int lbase = ((kl >> 4) << 4); // +M gives A lane
#pragma unroll
            for (int r = 0; r < 8; ++r) {
                int   M  = r + 8 * mhi;
                float iv = sigf(acc_i[r]);
                float fv = sigf(acc_f[r]);
                float gv = tanh_fast(acc_g[r]);
                float ov = sigf(acc_o[r]);
                float c  = fv * c_reg[q][r] + iv * gv;
                c_reg[q][r] = c;
                float h  = ov * tanh_fast(c);
                *(__bf16*)(hdst + (kt2 - 4) * 1024 + (lbase + M) * 32 + j * 2) = (__bf16)h;
                if (t == TT - 1)
                    hT[(size_t)(b0 + M) * HH + col] = h;
            }
        }
        __syncthreads();   // pong complete before next step stages/reads
    }
}

// ---------------------------------------------------------------------------
// Head: out[b] = sum_j W2[j] * relu(hT[b,:].W1[j,:] + b1[j]) + b2
// ---------------------------------------------------------------------------
__global__ __launch_bounds__(256) void head_kernel(
    const float* __restrict__ hT, const float* __restrict__ W1,
    const float* __restrict__ b1, const float* __restrict__ W2,
    const float* __restrict__ b2, float* __restrict__ out)
{
    __shared__ float hrow[HH];
    __shared__ float red[256];
    const int b   = blockIdx.x;
    const int tid = threadIdx.x;

    for (int k = tid; k < HH; k += 256) hrow[k] = hT[(size_t)b * HH + k];
    __syncthreads();

    float acc = 0.0f;
    for (int j = tid; j < 1024; j += 256) {
        const float* w = W1 + (size_t)j * HH;
        float d = b1[j];
#pragma unroll 8
        for (int k = 0; k < HH; ++k) d = fmaf(hrow[k], w[k], d);
        acc = fmaf(fmaxf(d, 0.0f), W2[j], acc);
    }
    red[tid] = acc;
    __syncthreads();
    for (int s = 128; s > 0; s >>= 1) {
        if (tid < s) red[tid] += red[tid + s];
        __syncthreads();
    }
    if (tid == 0) out[b] = red[0] + b2[0];
}

// ---------------------------------------------------------------------------
extern "C" void kernel_launch(void* const* d_in, const int* in_sizes, int n_in,
                              void* d_out, int out_size, void* d_ws, size_t ws_size,
                              hipStream_t stream)
{
    const float* x    = (const float*)d_in[0];
    const float* W_ih = (const float*)d_in[1];
    const float* W_hh = (const float*)d_in[2];
    const float* b_ih = (const float*)d_in[3];
    const float* b_hh = (const float*)d_in[4];
    const float* W1   = (const float*)d_in[5];
    const float* b1   = (const float*)d_in[6];
    const float* W2   = (const float*)d_in[7];
    const float* b2   = (const float*)d_in[8];

    char*  ws   = (char*)d_ws;
    float* hT   = (float*)(ws + HT_OFF);
    const __bf16* wp   = (const __bf16*)(ws + WP_OFF);
    const float*  bias = (const float*)(ws + BIAS_OFF);

    prep_kernel<<<1024, 256, 0, stream>>>(W_ih, W_hh, b_ih, b_hh, ws);
    lstm_kernel<<<BB / BTILE, 256, LDS_BYTES, stream>>>(x, wp, bias, hT);
    head_kernel<<<BB, 256, 0, stream>>>(hT, W1, b1, W2, b2, (float*)d_out);
}
